// SymmetricLoss_2911987827027
// MI455X (gfx1250) — matrix operand order repaired and checked by reading, not compiled, not run on hardware
//
#include <hip/hip_runtime.h>
#include <hip/hip_bf16.h>

typedef __attribute__((ext_vector_type(16))) _Float16 v16h;
typedef __attribute__((ext_vector_type(8)))  _Float16 v8h;
typedef __attribute__((ext_vector_type(4)))  _Float16 v4h;
typedef __attribute__((ext_vector_type(8)))  float    v8f;

#define NROWS   8192
#define DIM     512
#define TAU_INV 2.0f      // 1 / 0.5
#define BM      128
#define BN      128
#define BK      64
#define NCHUNK  (DIM / BK)   // 8
#define LDSX    72        // padded LDS row stride in halves (144 B = 9*16 -> frag loads stay 16B aligned)

// ---------------------------------------------------------------------------
// Kernel 0: zero the row/col exp-sum accumulators (deterministic per call)
// ---------------------------------------------------------------------------
__global__ void zero_f32_kernel(float* __restrict__ p, int n) {
  int i = blockIdx.x * blockDim.x + threadIdx.x;
  if (i < n) p[i] = 0.0f;
}

// ---------------------------------------------------------------------------
// Kernel 1: row L2-normalize fp32 -> fp16 (one wave32 per row, 8 rows / block)
// ---------------------------------------------------------------------------
__global__ __launch_bounds__(256)
void normalize_rows_kernel(const float* __restrict__ in, _Float16* __restrict__ out) {
  const int wave = threadIdx.x >> 5;
  const int lane = threadIdx.x & 31;
  const int row  = blockIdx.x * 8 + wave;

  const float4* rp = (const float4*)(in + (size_t)row * DIM);
  float4 v[4];
  float ss = 0.0f;
#pragma unroll
  for (int c = 0; c < 4; ++c) {                 // 32 lanes * 4 float4 = 512 floats
    v[c] = rp[lane + 32 * c];
    ss += v[c].x * v[c].x + v[c].y * v[c].y + v[c].z * v[c].z + v[c].w * v[c].w;
  }
#pragma unroll
  for (int off = 16; off >= 1; off >>= 1) ss += __shfl_xor(ss, off);

  const float s = 1.0f / fmaxf(sqrtf(ss), 1e-8f);   // torch CosineSimilarity eps

  v4h* op = (v4h*)(out + (size_t)row * DIM);
#pragma unroll
  for (int c = 0; c < 4; ++c) {
    v4h h;
    h[0] = (_Float16)(v[c].x * s);
    h[1] = (_Float16)(v[c].y * s);
    h[2] = (_Float16)(v[c].z * s);
    h[3] = (_Float16)(v[c].w * s);
    op[lane + 32 * c] = h;
  }
}

// ---------------------------------------------------------------------------
// Kernel 2: tiled WMMA GEMM + fused exp + row/col exp-sum + diagonal capture.
// Block = 256 threads = 8 waves laid out 4 (M) x 2 (N).
// Each wave owns a 32x64 patch: 2 M-subtiles x 4 N-subtiles (8 v8f accums).
//   -> per K-step: 4 A + 8 B ds_load_b128 feed 8 WMMAs (vs 18 loads before).
// Double-buffered LDS: next K-chunk's global loads issue while current chunk
// computes; staging regs drain into the alternate buffer (1 barrier / iter).
// Logits bounded by 1/tau = 2 -> softmax needs no max subtraction.
// ---------------------------------------------------------------------------
__global__ __launch_bounds__(256)
void cosine_gemm_exp_kernel(const _Float16* __restrict__ xn,
                            const _Float16* __restrict__ yn,
                            float* __restrict__ rowsum,
                            float* __restrict__ colsum,
                            float* __restrict__ diag) {
  __shared__ _Float16 Xs[2][BM * LDSX];
  __shared__ _Float16 Ys[2][BN * LDSX];

  const int tileM  = blockIdx.x * BM;
  const int tileN  = blockIdx.y * BN;
  const int tid    = threadIdx.x;
  const int wave   = tid >> 5;
  const int lane   = tid & 31;
  const int half16 = lane >> 4;     // 0: lanes 0-15, 1: lanes 16-31
  const int l16    = lane & 15;
  const int wr     = wave & 3;      // wave's M-pair index (rows wr*32 .. +32)
  const int wc     = wave >> 2;     // wave's N-quad index (cols wc*64 .. +64)

  v8f acc[2][4];
  {
    v8f z = {};
#pragma unroll
    for (int mi = 0; mi < 2; ++mi)
#pragma unroll
      for (int ni = 0; ni < 4; ++ni) acc[mi][ni] = z;
  }

  // cooperative loader: 8 threads per row (16B chunks), 4 rows per thread
  const int chunk = tid & 7;
  const int row0  = tid >> 3;       // 0..31

  const _Float16* gx = xn + (size_t)tileM * DIM + chunk * 8;
  const _Float16* gy = yn + (size_t)tileN * DIM + chunk * 8;

  float4 regX[4], regY[4];

  // ---- staging helpers (fully unrolled) ----
#define LOAD_REGS(kc)                                                          \
  _Pragma("unroll")                                                            \
  for (int rr = 0; rr < 4; ++rr) {                                             \
    const int r = row0 + rr * 32;                                              \
    regX[rr] = *(const float4*)(gx + (size_t)r * DIM + (kc) * BK);             \
    regY[rr] = *(const float4*)(gy + (size_t)r * DIM + (kc) * BK);             \
  }

#define STORE_LDS(buf)                                                         \
  _Pragma("unroll")                                                            \
  for (int rr = 0; rr < 4; ++rr) {                                             \
    const int r = row0 + rr * 32;                                              \
    *(float4*)(&Xs[(buf)][r * LDSX + chunk * 8]) = regX[rr];                   \
    *(float4*)(&Ys[(buf)][r * LDSX + chunk * 8]) = regY[rr];                   \
  }

  LOAD_REGS(0)
  STORE_LDS(0)

  for (int kc = 0; kc < NCHUNK; ++kc) {
    const int buf = kc & 1;
    __syncthreads();                       // buf is fully populated
    if (kc + 1 < NCHUNK) { LOAD_REGS(kc + 1) }   // overlap global with compute

#pragma unroll
    for (int kk = 0; kk < BK; kk += 32) {
      // A fragments (16x32 f16): lane hi=0 holds K{0-7,16-23}, hi=1 K{8-15,24-31}
      union { v16h v; v8h h[2]; } af[2];
#pragma unroll
      for (int mi = 0; mi < 2; ++mi) {
        const _Float16* p =
            &Xs[buf][(32 * wr + 16 * mi + l16) * LDSX + kk + half16 * 8];
        af[mi].h[0] = *(const v8h*)p;
        af[mi].h[1] = *(const v8h*)(p + 16);
      }
      // B fragments (32x16 f16): lane n = col n; lanes 0-15 K{0-7,16-23}, 16-31 K{8-15,24-31}
      union { v16h v; v8h h[2]; } bf[4];
#pragma unroll
      for (int ni = 0; ni < 4; ++ni) {
        const _Float16* p =
            &Ys[buf][(64 * wc + 16 * ni + l16) * LDSX + kk + half16 * 8];
        bf[ni].h[0] = *(const v8h*)p;
        bf[ni].h[1] = *(const v8h*)(p + 16);
      }
#pragma unroll
      for (int ni = 0; ni < 4; ++ni)
#pragma unroll
        for (int mi = 0; mi < 2; ++mi)
          acc[mi][ni] = __builtin_amdgcn_wmma_f32_16x16x32_f16(
              false, af[mi].v, false, bf[ni].v, (short)0, acc[mi][ni],
              false, false);
    }

    if (kc + 1 < NCHUNK) { STORE_LDS(buf ^ 1) }  // safe: other buffer
  }
#undef LOAD_REGS
#undef STORE_LDS

  // Epilogue. C-layout: VGPR r of lane l = element (row = r + 8*half16, col = l16).
  float rp[2][8];
#pragma unroll
  for (int mi = 0; mi < 2; ++mi)
#pragma unroll
    for (int r = 0; r < 8; ++r) rp[mi][r] = 0.0f;

#pragma unroll
  for (int ni = 0; ni < 4; ++ni) {
    const int gcol = tileN + 64 * wc + 16 * ni + l16;
    float csum = 0.0f;
#pragma unroll
    for (int mi = 0; mi < 2; ++mi) {
      const int growBase = tileM + 32 * wr + 16 * mi + 8 * half16;
#pragma unroll
      for (int r = 0; r < 8; ++r) {
        const float logit = acc[mi][ni][r] * TAU_INV;
        const float e = __expf(logit);
        rp[mi][r] += e;
        csum      += e;
        if (growBase + r == gcol) diag[gcol] = logit; // unique tile owns each diag
      }
    }
    csum += __shfl_xor(csum, 16);              // fold the two row-halves (same cols)
    if (half16 == 0) atomicAdd(&colsum[gcol], csum);
  }

#pragma unroll
  for (int mi = 0; mi < 2; ++mi) {
    const int growBase = tileM + 32 * wr + 16 * mi + 8 * half16;
#pragma unroll
    for (int r = 0; r < 8; ++r) {              // reduce 16 lanes of one row
      float v = rp[mi][r];
      v += __shfl_xor(v, 1);
      v += __shfl_xor(v, 2);
      v += __shfl_xor(v, 4);
      v += __shfl_xor(v, 8);
      if (l16 == 0) atomicAdd(&rowsum[growBase + r], v);
    }
  }
}

// ---------------------------------------------------------------------------
// Kernel 3: loss = mean_i [ 0.5*(log rowsum_i + log colsum_i) - diag_i ]
// ---------------------------------------------------------------------------
__global__ __launch_bounds__(256)
void final_loss_kernel(const float* __restrict__ rowsum,
                       const float* __restrict__ colsum,
                       const float* __restrict__ diag,
                       float* __restrict__ out) {
  __shared__ float red[256];
  float local = 0.0f;
  for (int i = threadIdx.x; i < NROWS; i += 256)
    local += 0.5f * (__logf(rowsum[i]) + __logf(colsum[i])) - diag[i];
  red[threadIdx.x] = local;
  __syncthreads();
  for (int s = 128; s > 0; s >>= 1) {
    if (threadIdx.x < s) red[threadIdx.x] += red[threadIdx.x + s];
    __syncthreads();
  }
  if (threadIdx.x == 0) out[0] = red[0] / (float)NROWS;
}

// ---------------------------------------------------------------------------
// Entry point
// ---------------------------------------------------------------------------
extern "C" void kernel_launch(void* const* d_in, const int* in_sizes, int n_in,
                              void* d_out, int out_size, void* d_ws, size_t ws_size,
                              hipStream_t stream) {
  const float* x = (const float*)d_in[0];   // [8192, 512] f32
  const float* y = (const float*)d_in[1];   // [8192, 512] f32

  // workspace layout: xn f16 (8MB) | yn f16 (8MB) | rowsum | colsum | diag (32KB each)
  char* ws = (char*)d_ws;
  _Float16* xn  = (_Float16*)ws;
  _Float16* yn  = (_Float16*)(ws + (size_t)NROWS * DIM * sizeof(_Float16));
  float* rowsum = (float*)(ws + (size_t)NROWS * DIM * sizeof(_Float16) * 2);
  float* colsum = rowsum + NROWS;
  float* diag   = colsum + NROWS;

  zero_f32_kernel<<<(2 * NROWS + 255) / 256, 256, 0, stream>>>(rowsum, 2 * NROWS);
  normalize_rows_kernel<<<NROWS / 8, 256, 0, stream>>>(x, xn);
  normalize_rows_kernel<<<NROWS / 8, 256, 0, stream>>>(y, yn);

  dim3 grid(NROWS / BM, NROWS / BN);
  cosine_gemm_exp_kernel<<<grid, 256, 0, stream>>>(xn, yn, rowsum, colsum, diag);

  final_loss_kernel<<<1, 256, 0, stream>>>(rowsum, colsum, diag, (float*)d_out);
}
